// ContMultiHeadedAttention_3161095930288
// MI455X (gfx1250) — matrix operand order repaired and checked by reading, not compiled, not run on hardware
//
#include <hip/hip_runtime.h>

// Problem constants (from reference)
#define NUM_HEADS 8
#define HEAD_DIM  64
#define EMB       512
#define SEQ       2048
#define BATCH     2
#define NTOK      (BATCH * SEQ)   // 4096

typedef __bf16          bf16x16 __attribute__((ext_vector_type(16)));
typedef float           f32x8   __attribute__((ext_vector_type(8)));

// Hardware f32 -> bf16 conversion (lowers to v_fma_mixlo_bf16 / native cvt)
__device__ __forceinline__ unsigned short f2bf(float f) {
  __bf16 b = (__bf16)f;
  return __builtin_bit_cast(unsigned short, b);
}

// ---- WMMA fragment loaders (layouts per CDNA5 ISA 7.12.2) ----
// A fragment (16x32 MxK): per-lane two contiguous 8-element chunks at p and p+16.
__device__ __forceinline__ bf16x16 ld_frag_a(const unsigned short* p) {
  union { uint4 q[2]; bf16x16 v; } u;
  u.q[0] = *(const uint4*)(p);
  u.q[1] = *(const uint4*)(p + 16);
  return u.v;
}
// B fragment where the 16 K-elements are contiguous in memory per lane.
__device__ __forceinline__ bf16x16 ld_frag_contig(const unsigned short* p) {
  union { uint4 q[2]; bf16x16 v; } u;
  u.q[0] = *(const uint4*)(p);
  u.q[1] = *(const uint4*)(p + 8);
  return u.v;
}

// -------- Preprocess: fp32 -> bf16 elementwise (activations) --------
__global__ __launch_bounds__(256) void cvt_kernel(const float* __restrict__ X,
                                                  unsigned short* __restrict__ Y) {
  int i = (blockIdx.x * 256 + threadIdx.x) * 8;
  float4 a = *(const float4*)(X + i);
  float4 b = *(const float4*)(X + i + 4);
  union { unsigned short s[8]; uint4 q; } o;
  o.s[0] = f2bf(a.x); o.s[1] = f2bf(a.y); o.s[2] = f2bf(a.z); o.s[3] = f2bf(a.w);
  o.s[4] = f2bf(b.x); o.s[5] = f2bf(b.y); o.s[6] = f2bf(b.z); o.s[7] = f2bf(b.w);
  *(uint4*)(Y + i) = o.q;
}

// -------- Preprocess: W[K][N] fp32 -> Wt[N][K] bf16 (LDS-tiled transpose) --------
__global__ __launch_bounds__(256) void wtrans_kernel(const float* __restrict__ W,
                                                     unsigned short* __restrict__ Wt) {
  __shared__ unsigned short tile[32][33];
  int tx = blockIdx.x & 15, ty = blockIdx.x >> 4;     // 16x16 tiles of 32x32
  int n = tx * 32 + threadIdx.x;
#pragma unroll
  for (int j = 0; j < 4; ++j) {
    int kl = threadIdx.y + j * 8;
    tile[kl][threadIdx.x] = f2bf(W[(size_t)(ty * 32 + kl) * EMB + n]);
  }
  __syncthreads();
  int k = ty * 32 + threadIdx.x;
#pragma unroll
  for (int j = 0; j < 4; ++j) {
    int nl = threadIdx.y + j * 8;
    Wt[(size_t)(tx * 32 + nl) * EMB + k] = tile[threadIdx.x][nl];
  }
}

// -------- Stage 1: projection. Y = (Xc @ Wt^T + bias) * scale, bf16 out --------
// Each wave computes a 16x64 slab: one shared A fragment, 4 accumulators.
// v_major == 0 : Y layout [B, H, S, 64]  (Q, K)
// v_major == 1 : Y layout [B, H, 64, S]  (V, d-major for contiguous ctx-B frags)
__global__ __launch_bounds__(256) void proj_kernel(const unsigned short* __restrict__ Xc,
                                                   const unsigned short* __restrict__ Wt,
                                                   const float* __restrict__ bias,
                                                   unsigned short* __restrict__ Y,
                                                   float scale, int v_major) {
  int lane = threadIdx.x;
  int wid  = blockIdx.x * 8 + threadIdx.y;        // 16x64 slab per wave
  int mt = wid >> 3, ng = wid & 7;                // EMB/64 = 8 column groups
  int row0 = mt * 16, col0 = ng * 64;
  int l16 = lane & 15, half = lane >> 4;
  int kb = half * 8, ks = half * 16;

  const unsigned short* ap = Xc + (size_t)(row0 + l16) * EMB;
  const unsigned short* bp = Wt + (size_t)(col0 + l16) * EMB;

  f32x8 zero = {};
  f32x8 acc[4];
#pragma unroll
  for (int t = 0; t < 4; ++t) acc[t] = zero;

#pragma unroll 4
  for (int k0 = 0; k0 < EMB; k0 += 32) {
    bf16x16 a = ld_frag_a(ap + k0 + kb);          // shared across 4 column tiles
#pragma unroll
    for (int t = 0; t < 4; ++t) {
      bf16x16 b = ld_frag_a(bp + (size_t)t * 16 * EMB + k0 + kb);
      acc[t] = __builtin_amdgcn_wmma_f32_16x16x32_bf16(false, a, false, b, (short)0,
                                                       acc[t], false, false);
    }
  }
#pragma unroll
  for (int t = 0; t < 4; ++t) {
    int c = col0 + t * 16 + l16;
    int h = c >> 6, d = c & 63;
    float bsv = bias[c];
#pragma unroll
    for (int r = 0; r < 8; ++r) {
      int tok = row0 + r + half * 8;
      int bb = tok >> 11, s = tok & (SEQ - 1);
      unsigned short v = f2bf((acc[t][r] + bsv) * scale);
      if (v_major)
        Y[((size_t)(bb * NUM_HEADS + h) * HEAD_DIM + d) * SEQ + s] = v;
      else
        Y[((size_t)(bb * NUM_HEADS + h) * SEQ + s) * HEAD_DIM + d] = v;
    }
  }
}

// -------- Stage 2: flash attention, fused spatial bias + boolean mask --------
__global__ __launch_bounds__(256) void attn_kernel(const unsigned short* __restrict__ Qp,
                                                   const unsigned short* __restrict__ Kp,
                                                   const unsigned short* __restrict__ Vt,
                                                   const float* __restrict__ sbias,
                                                   const unsigned char* __restrict__ mask,
                                                   unsigned short* __restrict__ Ctx) {
  __shared__ unsigned short plds[8][16 * 32];     // per-wave 16x32 prob tile (bf16 bits)
  int lane = threadIdx.x;
  int wv   = threadIdx.y;
  int wid  = blockIdx.x * 8 + wv;
  int qt = wid & 127;                              // SEQ/16 = 128 q tiles
  int h  = (wid >> 7) & 7;
  int b  = wid >> 10;
  int l16 = lane & 15, half = lane >> 4;
  int kb = half * 8, ks = half * 16;

  const unsigned short* Qb = Qp + (size_t)(b * NUM_HEADS + h) * SEQ * HEAD_DIM;
  const unsigned short* Kb = Kp + (size_t)(b * NUM_HEADS + h) * SEQ * HEAD_DIM;
  const unsigned short* Vb = Vt + (size_t)(b * NUM_HEADS + h) * HEAD_DIM * SEQ;
  const float*          Bb = sbias + (size_t)h * SEQ * SEQ;

  int qrow0 = qt * 16;
  const unsigned short* qrow = Qb + (size_t)(qrow0 + l16) * HEAD_DIM;
  bf16x16 aq0 = ld_frag_a(qrow + 0  + kb);        // d = 0..31
  bf16x16 aq1 = ld_frag_a(qrow + 32 + kb);        // d = 32..63

  f32x8 zero = {};
  f32x8 cacc[4];
#pragma unroll
  for (int t = 0; t < 4; ++t) cacc[t] = zero;
  float runmax[8], runsum[8];
  int   myrow[8];
#pragma unroll
  for (int r = 0; r < 8; ++r) {
    runmax[r] = -1e30f; runsum[r] = 0.f;
    myrow[r]  = qrow0 + r + half * 8;              // C/D layout: VGPR r -> rows r, r+8
  }

  for (int kb0 = 0; kb0 < SEQ; kb0 += 32) {
    // ---- two 16-key score subtiles, bias + mask fused in the C fragment ----
    f32x8 sm[2];
#pragma unroll
    for (int sub = 0; sub < 2; ++sub) {
      int key0 = kb0 + sub * 16;
      const unsigned short* krow = Kb + (size_t)(key0 + l16) * HEAD_DIM;
      bf16x16 bk0 = ld_frag_a(krow + 0  + kb);
      bf16x16 bk1 = ld_frag_a(krow + 32 + kb);
      f32x8 sc = zero;
      sc = __builtin_amdgcn_wmma_f32_16x16x32_bf16(false, aq0, false, bk0, (short)0, sc,
                                                   false, false);
      sc = __builtin_amdgcn_wmma_f32_16x16x32_bf16(false, aq1, false, bk1, (short)0, sc,
                                                   false, false);
      int col = key0 + l16;
#pragma unroll
      for (int r = 0; r < 8; ++r) {
        size_t off = (size_t)myrow[r] * SEQ + col;
        sm[sub][r] = mask[off] ? (sc[r] + Bb[off]) : -1e30f;
      }
    }
    // ---- one online-softmax reduction pass over all 32 keys ----
#pragma unroll
    for (int r = 0; r < 8; ++r) {
      float a = sm[0][r], bsc = sm[1][r];
      float tmax = fmaxf(a, bsc);
      tmax = fmaxf(tmax, __shfl_xor(tmax, 1));
      tmax = fmaxf(tmax, __shfl_xor(tmax, 2));
      tmax = fmaxf(tmax, __shfl_xor(tmax, 4));
      tmax = fmaxf(tmax, __shfl_xor(tmax, 8));
      float nmax = fmaxf(runmax[r], tmax);
      float corr = __expf(runmax[r] - nmax);
      runmax[r] = nmax;
      float p0 = __expf(a - nmax), p1 = __expf(bsc - nmax);
      float psum = p0 + p1;
      psum += __shfl_xor(psum, 1);
      psum += __shfl_xor(psum, 2);
      psum += __shfl_xor(psum, 4);
      psum += __shfl_xor(psum, 8);
      runsum[r] = runsum[r] * corr + psum;
#pragma unroll
      for (int t = 0; t < 4; ++t) cacc[t][r] = cacc[t][r] * corr;
      int rl = (r + half * 8) * 32 + l16;
      plds[wv][rl]      = f2bf(p0);
      plds[wv][rl + 16] = f2bf(p1);
    }
    __syncthreads();                                // uniform trip count across waves
    bf16x16 ap = ld_frag_a(&plds[wv][0] + (size_t)l16 * 32 + kb);
#pragma unroll
    for (int t = 0; t < 4; ++t) {
      // B[kk][d] from d-major V: two 8-key chunks at kb and kb+16 per lane
      bf16x16 bv = ld_frag_a(Vb + (size_t)(t * 16 + l16) * SEQ + kb0 + kb);
      cacc[t] = __builtin_amdgcn_wmma_f32_16x16x32_bf16(false, ap, false, bv, (short)0,
                                                        cacc[t], false, false);
    }
    __syncthreads();
  }

  // ctx -> [token, H*64] bf16 for the output projection
#pragma unroll
  for (int t = 0; t < 4; ++t) {
#pragma unroll
    for (int r = 0; r < 8; ++r) {
      float v = cacc[t][r] / runsum[r];
      Ctx[(size_t)(b * SEQ + myrow[r]) * EMB + h * HEAD_DIM + t * 16 + l16] = f2bf(v);
    }
  }
}

// -------- Stage 3: output projection, 16x64 slab per wave, fp32 out + bias --------
__global__ __launch_bounds__(256) void out_proj_kernel(const unsigned short* __restrict__ Ctx,
                                                       const unsigned short* __restrict__ Wt,
                                                       const float* __restrict__ bias,
                                                       float* __restrict__ Out) {
  int lane = threadIdx.x;
  int wid  = blockIdx.x * 8 + threadIdx.y;
  int mt = wid >> 3, ng = wid & 7;
  int row0 = mt * 16, col0 = ng * 64;
  int l16 = lane & 15, half = lane >> 4;
  int kb = half * 8, ks = half * 16;

  const unsigned short* ap = Ctx + (size_t)(row0 + l16) * EMB;
  const unsigned short* bp = Wt + (size_t)(col0 + l16) * EMB;

  f32x8 zero = {};
  f32x8 acc[4];
#pragma unroll
  for (int t = 0; t < 4; ++t) acc[t] = zero;

#pragma unroll 4
  for (int k0 = 0; k0 < EMB; k0 += 32) {
    bf16x16 a = ld_frag_a(ap + k0 + kb);
#pragma unroll
    for (int t = 0; t < 4; ++t) {
      bf16x16 b = ld_frag_a(bp + (size_t)t * 16 * EMB + k0 + kb);
      acc[t] = __builtin_amdgcn_wmma_f32_16x16x32_bf16(false, a, false, b, (short)0,
                                                       acc[t], false, false);
    }
  }
#pragma unroll
  for (int t = 0; t < 4; ++t) {
    int c = col0 + t * 16 + l16;
    float bsv = bias[c];
#pragma unroll
    for (int r = 0; r < 8; ++r)
      Out[(size_t)(row0 + r + half * 8) * EMB + c] = acc[t][r] + bsv;
  }
}

extern "C" void kernel_launch(void* const* d_in, const int* in_sizes, int n_in,
                              void* d_out, int out_size, void* d_ws, size_t ws_size,
                              hipStream_t stream) {
  // setup_inputs order: k, v, q, mask, spatial_bias, Wq, bq, Wk, bk, Wv, bv, Wo, bo
  const float*         k    = (const float*)d_in[0];
  const float*         v    = (const float*)d_in[1];
  const float*         q    = (const float*)d_in[2];
  const unsigned char* mask = (const unsigned char*)d_in[3];  // numpy bool_ = 1 byte
  const float*         sb   = (const float*)d_in[4];
  const float* Wq = (const float*)d_in[5];
  const float* bq = (const float*)d_in[6];
  const float* Wk = (const float*)d_in[7];
  const float* bk = (const float*)d_in[8];
  const float* Wv = (const float*)d_in[9];
  const float* bv = (const float*)d_in[10];
  const float* Wo = (const float*)d_in[11];
  const float* bo = (const float*)d_in[12];
  float* out = (float*)d_out;

  // Workspace carve-up (ushort units). Peak ~26 MB — resident in 192 MB L2.
  const size_t NX = (size_t)NTOK * EMB;   // 2M elems per activation buffer
  const size_t NW = (size_t)EMB * EMB;    // 256K elems per weight
  unsigned short* ws  = (unsigned short*)d_ws;
  unsigned short* Xq  = ws;               // converted q   (dead after proj)
  unsigned short* Xk  = Xq + NX;
  unsigned short* Xv  = Xk + NX;
  unsigned short* Wqt = Xv + NX;
  unsigned short* Wkt = Wqt + NW;
  unsigned short* Wvt = Wkt + NW;
  unsigned short* Wot = Wvt + NW;
  unsigned short* Qp  = Wot + NW;
  unsigned short* Kp  = Qp + NX;
  unsigned short* Vt  = Kp + NX;          // [B,H,64,S] d-major
  unsigned short* Cx  = Xq;               // alias: Xq dead once projections ran

  dim3 b256(256), b32x8(32, 8);
  int gcvt  = (int)(NX / (256 * 8));                // 1024
  int gtr   = 256;                                  // 16x16 tiles of 32x32
  int gproj = (NTOK / 16) * (EMB / 64) / 8;         // 256 (16x64 slabs)
  int gattn = BATCH * NUM_HEADS * (SEQ / 16) / 8;   // 256

  hipLaunchKernelGGL(cvt_kernel, dim3(gcvt), b256, 0, stream, q, Xq);
  hipLaunchKernelGGL(cvt_kernel, dim3(gcvt), b256, 0, stream, k, Xk);
  hipLaunchKernelGGL(cvt_kernel, dim3(gcvt), b256, 0, stream, v, Xv);
  hipLaunchKernelGGL(wtrans_kernel, dim3(gtr), b32x8, 0, stream, Wq, Wqt);
  hipLaunchKernelGGL(wtrans_kernel, dim3(gtr), b32x8, 0, stream, Wk, Wkt);
  hipLaunchKernelGGL(wtrans_kernel, dim3(gtr), b32x8, 0, stream, Wv, Wvt);
  hipLaunchKernelGGL(wtrans_kernel, dim3(gtr), b32x8, 0, stream, Wo, Wot);

  hipLaunchKernelGGL(proj_kernel, dim3(gproj), b32x8, 0, stream, Xq, Wqt, bq, Qp, 0.125f, 0);
  hipLaunchKernelGGL(proj_kernel, dim3(gproj), b32x8, 0, stream, Xk, Wkt, bk, Kp, 1.0f, 0);
  hipLaunchKernelGGL(proj_kernel, dim3(gproj), b32x8, 0, stream, Xv, Wvt, bv, Vt, 1.0f, 1);

  hipLaunchKernelGGL(attn_kernel, dim3(gattn), b32x8, 0, stream, Qp, Kp, Vt, sb, mask, Cx);
  hipLaunchKernelGGL(out_proj_kernel, dim3(gproj), b32x8, 0, stream, Cx, Wot, bo, out);
}
